// MoE_layer_35313221108105
// MI455X (gfx1250) — matrix operand order repaired and checked by reading, not compiled, not run on hardware
//
#include <hip/hip_runtime.h>
#include <hip/hip_bf16.h>
#include <math.h>

// ---------- types ----------
typedef __bf16 bf16_t;
typedef __attribute__((ext_vector_type(8)))  __bf16 v8bf;
typedef __attribute__((ext_vector_type(16))) __bf16 v16bf;
typedef __attribute__((ext_vector_type(8)))  float  v8f;

#define SHUF16(lo, hi) __builtin_shufflevector((lo), (hi), 0,1,2,3,4,5,6,7,8,9,10,11,12,13,14,15)

// ---------- problem constants ----------
#define BATCH 16
#define CIN   64
#define HID   128
#define HW    16384                       // 128*128
#define K1    576                         // 9 * CIN   (K order: tap-major, ci-minor)
#define K2    1152                        // 9 * HID
#define NEXP  8
#define Y_ELEMS  (BATCH * CIN * HW)       // 16,777,216
#define H_ELEMS  (HID * HW)               // 2,097,152 (one slot, NHWC bf16)

// ---------- workspace layout (bytes) ----------
#define XB_OFF    0u                      // x NHWC bf16: 33,554,432
#define W1_OFF    33554432u               // w1t bf16 [E][HID][9][CIN]: 1,179,648
#define W2_OFF    34734080u               // w2t bf16 [E][CIN][9][HID]: 1,179,648
#define AVG_OFF   35913728u
#define MAX_OFF   35917824u
#define SLOTE_OFF 35921920u
#define SLOTG_OFF 35922048u
#define H_BASE    35922432u               // h slots start (256-aligned)
#define HSLOT_B   4194304u                // bytes per h slot

// ---------- CDNA5 async global -> LDS helpers ----------
__device__ __forceinline__ unsigned lds_addr_of(const void* p) {
    // generic -> addrspace(3): numeric value is the LDS byte offset
    typedef __attribute__((address_space(3))) const char as3_char;
    return (unsigned)(unsigned long long)(as3_char*)p;
}

__device__ __forceinline__ void async_copy_b128(unsigned lds_byte_addr,
                                                unsigned gl_byte_off,
                                                unsigned long long gl_base) {
    // 16 bytes per active lane, no VGPR staging; tracked by ASYNCcnt
    asm volatile("global_load_async_to_lds_b128 %0, %1, %2"
                 :: "v"(lds_byte_addr), "v"(gl_byte_off), "s"(gl_base)
                 : "memory");
}

__device__ __forceinline__ void wait_async_all() {
    asm volatile("s_wait_asynccnt 0" ::: "memory");
}

// =====================================================================
// zero y
// =====================================================================
__global__ void moe_zero_y(float4* y) {
    int i = blockIdx.x * blockDim.x + threadIdx.x;
    y[i] = make_float4(0.f, 0.f, 0.f, 0.f);
}

// =====================================================================
// x: NCHW f32 -> NHWC bf16 via LDS transpose (64 ci x 64 px tiles)
// =====================================================================
__global__ void moe_xperm(const float* __restrict__ x, bf16_t* __restrict__ xb) {
    __shared__ bf16_t t[64][65];
    int b  = blockIdx.x >> 8;                 // 256 pixel tiles per batch
    int p0 = (blockIdx.x & 255) << 6;
    int tid = threadIdx.x;
    int q  = tid >> 6;                        // 0..3
    int pc = tid & 63;
    #pragma unroll
    for (int i = 0; i < 16; ++i) {
        int ci = q + i * 4;
        t[ci][pc] = (bf16_t)x[((size_t)(b * CIN + ci) << 14) + p0 + pc];
    }
    __syncthreads();
    #pragma unroll
    for (int i = 0; i < 16; ++i) {
        int pr = q + i * 4;
        xb[((size_t)((b << 14) + p0 + pr) << 6) + pc] = t[pc][pr];
    }
}

// =====================================================================
// weights: f32 [M][ci][3][3] -> bf16 [M][tap][ci]  (K reordered tap-major)
// =====================================================================
__global__ void moe_wtrans(const float* __restrict__ in, bf16_t* __restrict__ out,
                           int total, int cin) {
    int o = blockIdx.x * blockDim.x + threadIdx.x;
    if (o >= total) return;
    int ci   = o % cin;
    int rest = o / cin;
    int tap  = rest % 9;
    int mm   = rest / 9;
    out[o] = (bf16_t)in[((size_t)mm * cin + ci) * 9 + tap];
}

// =====================================================================
// avg + max pool; one block per (b,c)
// =====================================================================
__global__ void moe_pool(const float* __restrict__ x, float* __restrict__ avg,
                         float* __restrict__ mx) {
    int bc = blockIdx.x;
    const float* p = x + (size_t)bc * HW;
    float s = 0.f, m = -INFINITY;
    for (int i = threadIdx.x; i < HW; i += 256) {
        float v = p[i];
        s += v; m = fmaxf(m, v);
    }
    __shared__ float ss[256], sm[256];
    int t = threadIdx.x;
    ss[t] = s; sm[t] = m;
    __syncthreads();
    for (int off = 128; off > 0; off >>= 1) {
        if (t < off) { ss[t] += ss[t + off]; sm[t] = fmaxf(sm[t], sm[t + off]); }
        __syncthreads();
    }
    if (t == 0) { avg[bc] = ss[0] * (1.0f / HW); mx[bc] = sm[0]; }
}

// =====================================================================
// gating: logits -> top2 -> softmax -> slot table + CV^2 loss
// =====================================================================
__global__ void moe_gating(const float* __restrict__ avg, const float* __restrict__ mx,
                           const float* __restrict__ gw, const float* __restrict__ gb,
                           int* __restrict__ slot_e, float* __restrict__ slot_g,
                           float* __restrict__ loss_out) {
    __shared__ float g[BATCH][NEXP];
    int t = threadIdx.x;
    if (t < BATCH) {
        float logit[NEXP];
        for (int e = 0; e < NEXP; ++e) {
            float s = gb[e];
            const float* w = gw + e * (2 * CIN);
            for (int c = 0; c < CIN; ++c)
                s += avg[t * CIN + c] * w[c] + mx[t * CIN + c] * w[CIN + c];
            logit[e] = s;
        }
        int i1 = 0;
        for (int e = 1; e < NEXP; ++e) if (logit[e] > logit[i1]) i1 = e;
        int i2 = (i1 == 0) ? 1 : 0;
        for (int e = 0; e < NEXP; ++e)
            if (e != i1 && logit[e] > logit[i2]) i2 = e;
        float v1 = logit[i1], v2 = logit[i2];
        float m  = fmaxf(v1, v2);
        float e1 = expf(v1 - m), e2 = expf(v2 - m);
        float inv = 1.0f / (e1 + e2);
        float g1 = e1 * inv, g2 = e2 * inv;
        for (int e = 0; e < NEXP; ++e) g[t][e] = 0.f;
        g[t][i1] = g1; g[t][i2] = g2;
        slot_e[2 * t]     = i1; slot_g[2 * t]     = g1;
        slot_e[2 * t + 1] = i2; slot_g[2 * t + 1] = g2;
    }
    __syncthreads();
    if (t == 0) {
        float imp[NEXP], ld[NEXP];
        for (int e = 0; e < NEXP; ++e) {
            float si = 0.f, sl = 0.f;
            for (int b = 0; b < BATCH; ++b) {
                si += g[b][e];
                sl += (g[b][e] > 0.f) ? 1.f : 0.f;
            }
            imp[e] = si; ld[e] = sl;
        }
        float loss = 0.f;
        for (int which = 0; which < 2; ++which) {
            float* v = which ? ld : imp;
            float mean = 0.f;
            for (int e = 0; e < NEXP; ++e) mean += v[e];
            mean *= (1.0f / NEXP);
            float var = 0.f;
            for (int e = 0; e < NEXP; ++e) { float d = v[e] - mean; var += d * d; }
            var *= (1.0f / (NEXP - 1));
            loss += var / (mean * mean + 1e-10f);
        }
        loss_out[0] = loss * 0.01f;
    }
}

// =====================================================================
// conv1: implicit GEMM, bf16 WMMA, async-staged NHWC input row band.
// block = one image row (py) of one slot; LDS tile [3][130][64] bf16.
// wave owns pixel tile = wave id; loops 8 Cout tiles, 2 at a time
// (one B fragment feeds two WMMAs). Fused exact-erf GELU -> h NHWC bf16.
// =====================================================================
__global__ __launch_bounds__(256) void moe_conv1(
        const bf16_t* __restrict__ xb, const bf16_t* __restrict__ w1t,
        const int* __restrict__ slot_e, int base_b, int choice,
        bf16_t* __restrict__ hbuf) {
    __shared__ __align__(16) bf16_t tile[3 * 130 * CIN];      // 49,920 B
    int py = blockIdx.x;
    int ib = blockIdx.y;
    int b  = base_b + ib;
    int e  = slot_e[2 * b + choice];

    // ---- zero halo (whole tile), then async-fill valid interior rows ----
    v8bf* t8 = (v8bf*)tile;
    v8bf z = {};
    for (int i = threadIdx.x; i < 3 * 130 * CIN / 8; i += 256) t8[i] = z;
    __syncthreads();
    {
        unsigned ldsbase = lds_addr_of(tile);
        for (int row = 0; row < 3; ++row) {
            int yy = py - 1 + row;
            if ((unsigned)yy < 128u) {
                unsigned long long gb = (unsigned long long)(size_t)
                    (xb + (((size_t)(b << 7) + yy) << 7) * CIN);
                unsigned ldst = ldsbase + (unsigned)(row * (130 * CIN) + CIN) * 2u;
                for (int i = threadIdx.x; i < 128 * CIN / 8; i += 256)
                    async_copy_b128(ldst + (unsigned)i * 16u, (unsigned)i * 16u, gb);
            }
        }
        wait_async_all();
    }
    __syncthreads();

    int wave  = threadIdx.x >> 5;
    int lane  = threadIdx.x & 31;
    int khalf = lane >> 4;
    int mrow  = lane & 15;
    int col   = lane & 15;
    int px    = (wave << 4) + col;                 // pixel tile = wave
    int p     = (py << 7) + px;
    bf16_t* hs = hbuf + (size_t)ib * H_ELEMS;

    for (int mt = 0; mt < 8; mt += 2) {
        const bf16_t* A0 = w1t + ((size_t)(e * HID + (mt << 4) + mrow)) * K1;
        const bf16_t* A1 = A0 + 16 * K1;
        if (mt < 6) {                 // prefetch next Cout-tile pair's weights
            __builtin_prefetch(A0 + 32 * K1, 0, 0);
            __builtin_prefetch(A1 + 32 * K1, 0, 0);
        }
        v8f acc0 = {0.f,0.f,0.f,0.f,0.f,0.f,0.f,0.f};
        v8f acc1 = {0.f,0.f,0.f,0.f,0.f,0.f,0.f,0.f};
        for (int kk = 0; kk < K1; kk += 32) {
            int tap   = kk >> 6;                   // one filter tap per chunk
            int cbase = kk & 63;
            int dy = tap / 3, dx = tap - dy * 3;   // scalar per chunk
            int boff = (((dy * 130) + px + dx) << 6) + cbase + (khalf << 4);
            v8bf blo = *(const v8bf*)(tile + boff);
            v8bf bhi = *(const v8bf*)(tile + boff + 8);
            v16bf bfrag = SHUF16(blo, bhi);
            int aoff = kk + (khalf << 4);
            v16bf af0 = SHUF16(*(const v8bf*)(A0 + aoff),
                               *(const v8bf*)(A0 + aoff + 8));
            v16bf af1 = SHUF16(*(const v8bf*)(A1 + aoff),
                               *(const v8bf*)(A1 + aoff + 8));
            acc0 = __builtin_amdgcn_wmma_f32_16x16x32_bf16(
                false, af0, false, bfrag, (short)0, acc0, false, false);
            acc1 = __builtin_amdgcn_wmma_f32_16x16x32_bf16(
                false, af1, false, bfrag, (short)0, acc1, false, false);
        }
        // epilogue: GELU, NHWC store (8 contiguous channels per store)
        int mbase = (mt << 4) + (khalf << 3);
        v8bf o0, o1;
        #pragma unroll
        for (int r = 0; r < 8; ++r) {
            float v0 = acc0[r];
            o0[r] = (bf16_t)(0.5f * v0 * (1.0f + erff(v0 * 0.70710678118654752f)));
            float v1 = acc1[r];
            o1[r] = (bf16_t)(0.5f * v1 * (1.0f + erff(v1 * 0.70710678118654752f)));
        }
        *(v8bf*)(hs + ((size_t)p << 7) + mbase)      = o0;
        *(v8bf*)(hs + ((size_t)p << 7) + mbase + 16) = o1;
    }
}

// =====================================================================
// conv2: implicit GEMM, bf16 WMMA, LDS tile [3][66][128] (64-px half row),
// async-staged. Fused y += gate * o  (NCHW f32 output).
// =====================================================================
__global__ __launch_bounds__(256) void moe_conv2(
        const bf16_t* __restrict__ hbuf, const bf16_t* __restrict__ w2t,
        const int* __restrict__ slot_e, const float* __restrict__ slot_g,
        int base_b, int choice, float* __restrict__ y) {
    __shared__ __align__(16) bf16_t tile[3 * 66 * HID];       // 50,688 B
    int xhalf = blockIdx.x & 1;
    int py    = blockIdx.x >> 1;
    int px0   = xhalf << 6;
    int ib    = blockIdx.y;
    int b     = base_b + ib;
    int s     = 2 * b + choice;
    int e     = slot_e[s];
    float gate = slot_g[s];
    const bf16_t* hs = hbuf + (size_t)ib * H_ELEMS;

    // ---- zero, then async-fill interior + halo columns ----
    v8bf* t8 = (v8bf*)tile;
    v8bf z = {};
    for (int i = threadIdx.x; i < 3 * 66 * HID / 8; i += 256) t8[i] = z;
    __syncthreads();
    {
        unsigned ldsbase = lds_addr_of(tile);
        for (int row = 0; row < 3; ++row) {
            int yy = py - 1 + row;
            if ((unsigned)yy < 128u) {
                unsigned long long gb = (unsigned long long)(size_t)
                    (hs + (((size_t)(yy << 7)) + px0) * HID);
                unsigned ldst = ldsbase + (unsigned)(row * (66 * HID) + HID) * 2u;
                for (int i = threadIdx.x; i < 64 * HID / 8; i += 256)
                    async_copy_b128(ldst + (unsigned)i * 16u, (unsigned)i * 16u, gb);
                // halo columns (16 x b128 each side), regular path
                int t = threadIdx.x;
                if (t < 16 && px0 > 0) {
                    ((v8bf*)(tile + (row * 66) * HID))[t] =
                        ((const v8bf*)(hs + (((size_t)(yy << 7)) + px0 - 1) * HID))[t];
                } else if (t >= 16 && t < 32 && px0 + 64 < 128) {
                    ((v8bf*)(tile + (row * 66 + 65) * HID))[t - 16] =
                        ((const v8bf*)(hs + (((size_t)(yy << 7)) + px0 + 64) * HID))[t - 16];
                }
            }
        }
        wait_async_all();
    }
    __syncthreads();

    int wave  = threadIdx.x >> 5;
    int lane  = threadIdx.x & 31;
    int khalf = lane >> 4;
    int mrow  = lane & 15;
    int col   = lane & 15;
    int ptile = wave & 3;
    int mtp   = wave >> 2;                       // 0/1 -> Cout tiles {0,1}/{2,3}
    int pxl   = (ptile << 4) + col;              // local pixel 0..63
    int p     = (py << 7) + px0 + pxl;

    int mt = mtp << 1;
    const bf16_t* A0 = w2t + ((size_t)(e * CIN + (mt << 4) + mrow)) * K2;
    const bf16_t* A1 = A0 + 16 * K2;
    v8f acc0 = {0.f,0.f,0.f,0.f,0.f,0.f,0.f,0.f};
    v8f acc1 = {0.f,0.f,0.f,0.f,0.f,0.f,0.f,0.f};
    for (int kk = 0; kk < K2; kk += 32) {
        int tap   = kk >> 7;
        int cbase = kk & 127;
        int dy = tap / 3, dx = tap - dy * 3;
        int boff = (dy * 66 + pxl + dx) * HID + cbase + (khalf << 4);
        v8bf blo = *(const v8bf*)(tile + boff);
        v8bf bhi = *(const v8bf*)(tile + boff + 8);
        v16bf bfrag = SHUF16(blo, bhi);
        int aoff = kk + (khalf << 4);
        v16bf af0 = SHUF16(*(const v8bf*)(A0 + aoff),
                           *(const v8bf*)(A0 + aoff + 8));
        v16bf af1 = SHUF16(*(const v8bf*)(A1 + aoff),
                           *(const v8bf*)(A1 + aoff + 8));
        acc0 = __builtin_amdgcn_wmma_f32_16x16x32_bf16(
            false, af0, false, bfrag, (short)0, acc0, false, false);
        acc1 = __builtin_amdgcn_wmma_f32_16x16x32_bf16(
            false, af1, false, bfrag, (short)0, acc1, false, false);
    }
    int m0 = (mt << 4) + (khalf << 3);
    float* Yb = y + ((size_t)b << 20);           // b * 64 * 16384
    #pragma unroll
    for (int r = 0; r < 8; ++r) {
        size_t i0 = ((size_t)(m0 + r) << 14) + p;
        Yb[i0] += gate * acc0[r];
        size_t i1 = ((size_t)(m0 + 16 + r) << 14) + p;
        Yb[i1] += gate * acc1[r];
    }
}

// =====================================================================
// host-side launch
// =====================================================================
extern "C" void kernel_launch(void* const* d_in, const int* in_sizes, int n_in,
                              void* d_out, int out_size, void* d_ws, size_t ws_size,
                              hipStream_t stream) {
    (void)in_sizes; (void)n_in; (void)out_size;
    const float* x  = (const float*)d_in[0];
    const float* w1 = (const float*)d_in[1];
    const float* w2 = (const float*)d_in[2];
    const float* gw = (const float*)d_in[3];
    const float* gb = (const float*)d_in[4];
    float* y = (float*)d_out;

    char* ws = (char*)d_ws;
    bf16_t* xb   = (bf16_t*)(ws + XB_OFF);
    bf16_t* w1t  = (bf16_t*)(ws + W1_OFF);
    bf16_t* w2t  = (bf16_t*)(ws + W2_OFF);
    float*  avg  = (float*) (ws + AVG_OFF);
    float*  mx   = (float*) (ws + MAX_OFF);
    int*    se   = (int*)   (ws + SLOTE_OFF);
    float*  sg   = (float*) (ws + SLOTG_OFF);
    bf16_t* hbuf = (bf16_t*)(ws + H_BASE);

    // how many h slots fit in workspace (deterministic per environment)
    size_t avail = (ws_size > (size_t)H_BASE) ? (ws_size - H_BASE) / HSLOT_B : 1;
    int nbuf = (int)(avail < 1 ? 1 : (avail > 16 ? 16 : avail));

    // 1. zero y
    moe_zero_y<<<Y_ELEMS / 4 / 256, 256, 0, stream>>>((float4*)y);

    // 2. layout conversions
    moe_xperm<<<BATCH * 256, 256, 0, stream>>>(x, xb);
    const int W1N = NEXP * HID * 9 * CIN;
    const int W2N = NEXP * CIN * 9 * HID;
    moe_wtrans<<<(W1N + 255) / 256, 256, 0, stream>>>(w1, w1t, W1N, CIN);
    moe_wtrans<<<(W2N + 255) / 256, 256, 0, stream>>>(w2, w2t, W2N, HID);

    // 3. gating
    moe_pool<<<BATCH * CIN, 256, 0, stream>>>(x, avg, mx);
    moe_gating<<<1, 32, 0, stream>>>(avg, mx, gw, gb, se, sg, y + Y_ELEMS);

    // 4. routed experts. Groups contain at most one slot per batch sample
    //    (choice-major), so conv2's y += never races within a launch.
    for (int choice = 0; choice < 2; ++choice) {
        for (int b0 = 0; b0 < BATCH; b0 += nbuf) {
            int cnt = (BATCH - b0) < nbuf ? (BATCH - b0) : nbuf;
            moe_conv1<<<dim3(128, cnt), 256, 0, stream>>>(xb, w1t, se, b0, choice, hbuf);
            moe_conv2<<<dim3(256, cnt), 256, 0, stream>>>(hbuf, w2t, se, sg, b0, choice, y);
        }
    }
}
